// Transformer_14405320311120
// MI455X (gfx1250) — compile-verified, matrix operand order checked by reading, not run on hardware
//
#include <hip/hip_runtime.h>
#include <stdint.h>

// ---------------- problem constants ----------------
constexpr int cB  = 4;
constexpr int cT  = 1024;
constexpr int cD  = 1024;
constexpr int cH  = 16;
constexpr int cF  = 4096;
constexpr int cL  = 8;
constexpr int cNF = 80;
constexpr int cE  = 256;
constexpr int cG  = 16;
constexpr int cDh = 64;
constexpr int cCin = 2 * cNF + cE;     // 416
constexpr int cBT  = cB * cT;          // 4096
constexpr int CONV_K = 128;

// ---------------- vector types ----------------
typedef __attribute__((ext_vector_type(16))) __bf16 v16bf;
typedef __attribute__((ext_vector_type(8)))  __bf16 v8bf;
typedef __attribute__((ext_vector_type(8)))  float  v8f;
typedef __attribute__((ext_vector_type(4)))  float  v4f;
typedef __attribute__((ext_vector_type(4)))  int    v4i;
typedef __attribute__((address_space(1))) v4i* gv4i_t;  // global int4*
typedef __attribute__((address_space(3))) v4i* lv4i_t;  // LDS int4*

// ---------------- helpers ----------------
__device__ inline uint16_t f2bfu(float f) {
  uint32_t u = __builtin_bit_cast(uint32_t, f);
  u += 0x7fffu + ((u >> 16) & 1u);          // round-to-nearest-even
  return (uint16_t)(u >> 16);
}
__device__ inline __bf16 f2bf(float f) {
  uint16_t h = f2bfu(f);
  return __builtin_bit_cast(__bf16, h);
}
__device__ inline float gelu_exact(float x) {
  return 0.5f * x * (1.0f + erff(x * 0.70710678118654752f));
}
__device__ inline v8f wmma_bf16(v16bf a, v16bf b, v8f c) {
  return __builtin_amdgcn_wmma_f32_16x16x32_bf16(false, a, false, b, (short)0, c, false, false);
}
// A/B 16-bit fragment (ISA 7.12.2): lane holds 8 contiguous k at base, 8 more at base+16.
__device__ inline v16bf load_bf_frag(const uint16_t* base) {
  v8bf lo = *(const v8bf*)base;
  v8bf hi = *(const v8bf*)(base + 16);
  v16bf r;
#pragma unroll
  for (int e = 0; e < 8; ++e) { r[e] = lo[e]; r[8 + e] = hi[e]; }
  return r;
}

// ---- CDNA5 async global->LDS copy (ASYNCcnt-tracked), with sync fallback ----
#if __has_builtin(__builtin_amdgcn_global_load_async_to_lds_b128) && \
    __has_builtin(__builtin_amdgcn_s_wait_asynccnt)
#define HAVE_ASYNC_LDS 1
#else
#define HAVE_ASYNC_LDS 0
#endif

__device__ inline void cp16_g2l(const uint16_t* g, uint16_t* l) {
#if HAVE_ASYNC_LDS
  __builtin_amdgcn_global_load_async_to_lds_b128(
      (gv4i_t)(v4i*)(void*)g, (lv4i_t)(v4i*)(void*)l, 0, 0);
#else
  *(v8bf*)l = *(const v8bf*)g;
#endif
}
__device__ inline void wait_tile_inflight() {   // leave next tile's 3 ops outstanding
#if HAVE_ASYNC_LDS
  __builtin_amdgcn_s_wait_asynccnt(3);
#endif
}
__device__ inline void wait_tile_all() {
#if HAVE_ASYNC_LDS
  __builtin_amdgcn_s_wait_asynccnt(0);
#endif
}

// ---------------- fp32 -> bf16 bulk conversion (weights, once per launch) ----
__global__ void k_cvt(const float* __restrict__ s, uint16_t* __restrict__ d, int n) {
  for (int i = blockIdx.x * blockDim.x + threadIdx.x; i < n; i += gridDim.x * blockDim.x)
    d[i] = f2bfu(s[i]);
}

// ---------------- WMMA GEMM, double-buffered async LDS staging ----------------
// C[M,N] = act((A[M,K]_bf16 * W[N,K]_bf16^T + bias) * alpha)
// block tile 64x128, BK=32; 8 waves, each wave = 16x64 strip (4 accumulators).
__global__ __launch_bounds__(256) void k_gemm(
    const uint16_t* __restrict__ A, const uint16_t* __restrict__ W,
    const float* __restrict__ bias, float* __restrict__ Cf,
    uint16_t* __restrict__ Cbf, int M, int N, int Kd, float alpha, int act) {
  __shared__ uint16_t Asm[2][64 * 32];    // 8 KB
  __shared__ uint16_t Bsm[2][128 * 32];   // 16 KB
  const int tid = threadIdx.x;
  const int w = tid >> 5, lane = tid & 31;
  const int lh = lane >> 4, l16 = lane & 15;
  const int mblk = blockIdx.x * 64;
  const int nblk = blockIdx.y * 128;
  const int mi = w & 3, nj0 = (w >> 2) * 4;

  // staging assignment: A = 64 rows x 4 x16B chunks (1/thread); B = 128 x 4 (2/thread)
  const int arow = tid >> 2, achk = tid & 3;
  const int br0 = (tid * 2) >> 2, bc0 = (tid * 2) & 3;
  const int br1 = (tid * 2 + 1) >> 2, bc1 = (tid * 2 + 1) & 3;
  const uint16_t* Ag = A + (size_t)(mblk + arow) * Kd + achk * 8;
  int bn0 = nblk + br0; if (bn0 >= N) bn0 = nblk;
  int bn1 = nblk + br1; if (bn1 >= N) bn1 = nblk;
  const uint16_t* Bg0 = W + (size_t)bn0 * Kd + bc0 * 8;
  const uint16_t* Bg1 = W + (size_t)bn1 * Kd + bc1 * 8;
  const int aoff = arow * 32 + achk * 8;
  const int boff0 = br0 * 32 + bc0 * 8;
  const int boff1 = br1 * 32 + bc1 * 8;

  auto issue = [&](int ks, int buf) {
    cp16_g2l(Ag + ks, &Asm[buf][aoff]);
    cp16_g2l(Bg0 + ks, &Bsm[buf][boff0]);
    cp16_g2l(Bg1 + ks, &Bsm[buf][boff1]);
  };

  const int nk = Kd / 32;
  issue(0, 0);
  if (nk > 1) issue(32, 1);

  v8f zero = {};
  v8f acc[4] = {zero, zero, zero, zero};
  for (int i = 0; i < nk; ++i) {
    if (i + 1 < nk) wait_tile_inflight(); else wait_tile_all();
    __syncthreads();
    const int buf = i & 1;
    v16bf af = load_bf_frag(&Asm[buf][(mi * 16 + l16) * 32 + lh * 8]);
#pragma unroll
    for (int j = 0; j < 4; ++j) {
      v16bf bf_ = load_bf_frag(&Bsm[buf][((nj0 + j) * 16 + l16) * 32 + lh * 8]);
      acc[j] = wmma_bf16(af, bf_, acc[j]);
    }
    __syncthreads();                       // all waves done reading buf before refill
    if (i + 2 < nk) issue((i + 2) * 32, buf);
  }

#pragma unroll
  for (int j = 0; j < 4; ++j) {
    const int n = nblk + (nj0 + j) * 16 + l16;
    if (n >= N) continue;
    const float bs = bias ? bias[n] : 0.0f;
#pragma unroll
    for (int r = 0; r < 8; ++r) {
      int m = mblk + mi * 16 + r + lh * 8;
      float x = (acc[j][r] + bs) * alpha;
      if (act == 1) x = gelu_exact(x);
      size_t idx = (size_t)m * N + n;
      if (Cf) Cf[idx] = x;
      if (Cbf) Cbf[idx] = f2bfu(x);
    }
  }
}

// ---------------- attention ----------------
// block = (b, head, 64-row q tile). S[64][T] fp32 in dynamic LDS (stride 1028).
constexpr int SW = cT + 4;
__global__ __launch_bounds__(256) void k_attn(
    const uint16_t* __restrict__ Qbf, const uint16_t* __restrict__ Kbf,
    const uint16_t* __restrict__ VTbf, const float* __restrict__ ymask,
    uint16_t* __restrict__ Obf) {
  extern __shared__ float S[];
  const int blk = blockIdx.x;
  const int qt = blk & 15;            // T/64 tiles
  const int bh = blk >> 4;
  const int b = bh >> 4, hh = bh & 15;
  const int tid = threadIdx.x;
  const int w = tid >> 5, lane = tid & 31;
  const int lh = lane >> 4, l16 = lane & 15;
  const int mi = w & 3, ni0 = (w >> 2) * 2;
  const float slope = exp2f(-0.5f * (float)(hh + 1));

  const uint16_t* Qb = Qbf + (size_t)bh * cT * cDh;
  const uint16_t* Kb = Kbf + (size_t)bh * cT * cDh;
  const uint16_t* Vb = VTbf + (size_t)bh * cDh * cT;

  const int m0 = mi * 16;
  const int qrow = qt * 64 + m0 + l16;
  v16bf qfrag[2];
#pragma unroll
  for (int kc = 0; kc < 2; ++kc)
    qfrag[kc] = load_bf_frag(Qb + (size_t)qrow * cDh + kc * 32 + lh * 8);

  // phase 1: S = Q K^T + alibi + key mask
  for (int kt = 0; kt < 16; ++kt) {
    const int kbase = kt * 64;
#pragma unroll
    for (int nj = 0; nj < 2; ++nj) {
      const int kcol = kbase + (ni0 + nj) * 16 + l16;
      v8f acc = {};
#pragma unroll
      for (int kc = 0; kc < 2; ++kc) {
        v16bf kf = load_bf_frag(Kb + (size_t)kcol * cDh + kc * 32 + lh * 8);
        acc = wmma_bf16(qfrag[kc], kf, acc);
      }
      const float mterm = (1.0f - ymask[b * cT + kcol]) * -1.0e9f;
#pragma unroll
      for (int r = 0; r < 8; ++r) {
        int ml = m0 + r + lh * 8;
        int qg = qt * 64 + ml;
        S[ml * SW + kcol] = acc[r] - slope * fabsf((float)(qg - kcol)) + mterm;
      }
    }
  }
  __syncthreads();

  // phase 2: softmax, 8 rows per wave
#pragma unroll 1
  for (int r = 0; r < 8; ++r) {
    float* Sr = S + (w * 8 + r) * SW;
    float mx = -1.0e30f;
    for (int c = lane; c < cT; c += 32) mx = fmaxf(mx, Sr[c]);
    for (int mm = 16; mm; mm >>= 1) mx = fmaxf(mx, __shfl_xor(mx, mm, 32));
    float sum = 0.0f;
    for (int c = lane; c < cT; c += 32) { float e = __expf(Sr[c] - mx); Sr[c] = e; sum += e; }
    for (int mm = 16; mm; mm >>= 1) sum += __shfl_xor(sum, mm, 32);
    const float inv = 1.0f / sum;
    for (int c = lane; c < cT; c += 32) Sr[c] *= inv;
  }
  __syncthreads();

  // phase 3: O = P V   (V pre-transposed -> contiguous B-operand loads)
  v8f o0 = {}, o1 = {};
  const int arow = m0 + l16;
  for (int kt = 0; kt < 16; ++kt) {
#pragma unroll
    for (int kc = 0; kc < 2; ++kc) {
      const int kof = kt * 64 + kc * 32 + lh * 8;
      const float* ps = S + arow * SW + kof;
      v4f p0 = *(const v4f*)ps, p1 = *(const v4f*)(ps + 4);
      v4f p2 = *(const v4f*)(ps + 16), p3 = *(const v4f*)(ps + 20);
      v16bf pf;
#pragma unroll
      for (int e = 0; e < 4; ++e) {
        pf[e] = f2bf(p0[e]); pf[4 + e] = f2bf(p1[e]);
        pf[8 + e] = f2bf(p2[e]); pf[12 + e] = f2bf(p3[e]);
      }
      const int d0 = ni0 * 16 + l16, d1 = d0 + 16;
      v16bf vf0 = load_bf_frag(Vb + (size_t)d0 * cT + kof);
      o0 = wmma_bf16(pf, vf0, o0);
      v16bf vf1 = load_bf_frag(Vb + (size_t)d1 * cT + kof);
      o1 = wmma_bf16(pf, vf1, o1);
    }
  }
#pragma unroll
  for (int r = 0; r < 8; ++r) {
    int ml = m0 + r + lh * 8;
    int t = qt * 64 + ml;
    size_t base = ((size_t)(b * cT + t)) * cD + hh * cDh;
    Obf[base + ni0 * 16 + l16] = f2bfu(o0[r]);
    Obf[base + ni0 * 16 + 16 + l16] = f2bfu(o1[r]);
  }
}

// ---------------- fused add + mask + layernorm ----------------
// flags: 1 = add badd, 2 = mask before LN, 4 = mask after LN
__global__ __launch_bounds__(256) void k_ln(
    const float* __restrict__ a, const float* __restrict__ badd,
    const float* __restrict__ ymask, const float* __restrict__ g,
    const float* __restrict__ be, float* __restrict__ outf,
    uint16_t* __restrict__ outbf, uint16_t* __restrict__ skipbf, int flags) {
  __shared__ float red[256];
  const int row = blockIdx.x, tid = threadIdx.x;
  const float mk = ymask[row];
  const float mb = (flags & 2) ? mk : 1.0f;
  float x[4];
  float s = 0.0f;
#pragma unroll
  for (int j = 0; j < 4; ++j) {
    int c = tid + j * 256;
    float v = a[(size_t)row * cD + c];
    if (flags & 1) v += badd[(size_t)row * cD + c];
    v *= mb;
    x[j] = v; s += v;
  }
  red[tid] = s; __syncthreads();
  for (int st = 128; st; st >>= 1) { if (tid < st) red[tid] += red[tid + st]; __syncthreads(); }
  const float mean = red[0] * (1.0f / cD);
  __syncthreads();
  float sq = 0.0f;
#pragma unroll
  for (int j = 0; j < 4; ++j) { float d2 = x[j] - mean; sq += d2 * d2; }
  red[tid] = sq; __syncthreads();
  for (int st = 128; st; st >>= 1) { if (tid < st) red[tid] += red[tid + st]; __syncthreads(); }
  const float inv = rsqrtf(red[0] * (1.0f / cD) + 1e-5f);
  const float ma = (flags & 4) ? mk : 1.0f;
#pragma unroll
  for (int j = 0; j < 4; ++j) {
    int c = tid + j * 256;
    float v = ((x[j] - mean) * inv * g[c] + be[c]) * ma;
    outf[(size_t)row * cD + c] = v;
    uint16_t hb = f2bfu(v);
    outbf[(size_t)row * cD + c] = hb;
    if (skipbf) skipbf[(size_t)row * cD + c] = hb;
  }
}

// ---------------- elementwise / prep kernels ----------------
__global__ void k_build_hin(const float* __restrict__ x, const float* __restrict__ mu,
                            const int* __restrict__ tok, const float* __restrict__ emb,
                            uint16_t* __restrict__ out) {
  const int total = cBT * cCin;
  for (int i = blockIdx.x * blockDim.x + threadIdx.x; i < total; i += gridDim.x * blockDim.x) {
    int row = i / cCin, c = i - row * cCin;
    int b = row >> 10, t = row & 1023;
    float v;
    if (c < cNF)            v = x[((size_t)b * cNF + c) * cT + t];
    else if (c < 2 * cNF)   v = mu[((size_t)b * cNF + (c - cNF)) * cT + t];
    else                    v = emb[(size_t)tok[row] * cE + (c - 2 * cNF)];
    out[i] = f2bfu(v);
  }
}

__global__ void k_temb(float* __restrict__ h, const float* __restrict__ tv,
                       const float* __restrict__ ymask) {
  const int total = cBT * cD;
  for (int i = blockIdx.x * blockDim.x + threadIdx.x; i < total; i += gridDim.x * blockDim.x) {
    int d = i & 1023, row = i >> 10, b = row >> 10;
    int j = d & 511;
    float freq = __expf((float)j * (-9.210340371976184f / 511.0f));
    float ang = 1000.0f * tv[b] * freq;
    float te = (d < 512) ? __sinf(ang) : __cosf(ang);
    h[i] = (h[i] + te) * ymask[row];
  }
}

__global__ void k_add(float* __restrict__ h, const float* __restrict__ p) {
  const int total = cBT * cD;
  for (int i = blockIdx.x * blockDim.x + threadIdx.x; i < total; i += gridDim.x * blockDim.x)
    h[i] += p[i];
}

// grouped conv, one block = (b, group, 32-t tile); LDS tile channel-major (conflict free)
__global__ __launch_bounds__(256) void k_conv(
    const float* __restrict__ h, const float* __restrict__ wgt,
    const float* __restrict__ bias, const float* __restrict__ ymask,
    float* __restrict__ outp) {
  __shared__ float tile[64 * 160];   // [c][t-extent 159 (+pad)]
  const int blk = blockIdx.x;
  const int tt = blk & 31, g = (blk >> 5) & 15, b = blk >> 9;
  const int t0 = tt * 32;
  for (int i = threadIdx.x; i < 64 * 159; i += 256) {
    int c = i / 159, trow = i - c * 159;
    int t = t0 - 64 + trow;
    tile[c * 160 + trow] = (t >= 0 && t < cT)
        ? h[((size_t)b * cT + t) * cD + g * 64 + c] : 0.0f;
  }
  __syncthreads();
  const int tl = threadIdx.x & 31;
  const int oc0 = threadIdx.x >> 5;
  const int t = t0 + tl;
  const float mk = ymask[b * cT + t];
#pragma unroll 1
  for (int os = 0; os < 8; ++os) {
    int oc = oc0 * 8 + os;
    int d = g * 64 + oc;
    const float* wd = wgt + (size_t)d * 64 * CONV_K;
    float acc = bias[d];
#pragma unroll 1
    for (int c = 0; c < 64; ++c) {
      const float* wc = wd + c * CONV_K;
      const float* tc = tile + c * 160 + tl;
#pragma unroll 4
      for (int k = 0; k < CONV_K; ++k) acc += tc[k] * wc[k];
    }
    outp[((size_t)b * cT + t) * cD + d] = gelu_exact(acc) * mk;
  }
}

__global__ void k_pack_qkv(const float* __restrict__ qf, const float* __restrict__ kf,
                           const float* __restrict__ vf, uint16_t* __restrict__ Q,
                           uint16_t* __restrict__ Kq, uint16_t* __restrict__ VT) {
  const int total = cBT * cD;
  for (int i = blockIdx.x * blockDim.x + threadIdx.x; i < total; i += gridDim.x * blockDim.x) {
    int d = i & 1023, t = (i >> 10) & 1023, b = i >> 20;
    int hh = d >> 6, dh = d & 63;
    size_t bh = (size_t)(b * cH + hh);
    Q[(bh * cT + t) * cDh + dh] = f2bfu(qf[i]);
    Kq[(bh * cT + t) * cDh + dh] = f2bfu(kf[i]);
    VT[(bh * cDh + dh) * cT + t] = f2bfu(vf[i]);
  }
}

__global__ void k_concat(const uint16_t* __restrict__ hbf, const uint16_t* __restrict__ sbf,
                         uint16_t* __restrict__ out) {
  const int total = cBT * 2 * cD;
  for (int i = blockIdx.x * blockDim.x + threadIdx.x; i < total; i += gridDim.x * blockDim.x) {
    int row = i >> 11, c = i & 2047;
    out[i] = (c < cD) ? hbf[((size_t)row << 10) + c] : sbf[((size_t)row << 10) + (c - cD)];
  }
}

__global__ void k_outT(const float* __restrict__ tmp, const float* __restrict__ ymask,
                       float* __restrict__ out) {
  const int total = cB * cNF * cT;
  for (int i = blockIdx.x * blockDim.x + threadIdx.x; i < total; i += gridDim.x * blockDim.x) {
    int t = i & 1023;
    int f = (i >> 10) % cNF;
    int b = i / (cNF << 10);
    out[i] = tmp[((size_t)(b << 10) + t) * cNF + f] * ymask[(b << 10) + t];
  }
}

// ---------------- host driver ----------------
extern "C" void kernel_launch(void* const* d_in, const int* in_sizes, int n_in,
                              void* d_out, int out_size, void* d_ws, size_t ws_size,
                              hipStream_t stream) {
  (void)in_sizes; (void)n_in; (void)out_size; (void)ws_size;
  const float* x = (const float*)d_in[0];
  const float* mu = (const float*)d_in[1];
  const int* tokens = (const int*)d_in[2];
  const float* tvals = (const float*)d_in[3];
  const float* ymask = (const float*)d_in[4];
  const float* emb = (const float*)d_in[5];
  const float* proj_in_w = (const float*)d_in[6];
  const float* proj_in_b = (const float*)d_in[7];
  const float* posconv_w = (const float*)d_in[8];
  const float* posconv_b = (const float*)d_in[9];
  const float* ln0_g = (const float*)d_in[10];
  const float* ln0_b = (const float*)d_in[11];
  const float* qw = (const float*)d_in[12];
  const float* qb = (const float*)d_in[13];
  const float* kw = (const float*)d_in[14];
  const float* kb = (const float*)d_in[15];
  const float* vw = (const float*)d_in[16];
  const float* vb = (const float*)d_in[17];
  const float* ow = (const float*)d_in[18];
  const float* ob = (const float*)d_in[19];
  const float* ln1_g = (const float*)d_in[20];
  const float* ln1_b = (const float*)d_in[21];
  const float* ff_w1 = (const float*)d_in[22];
  const float* ff_b1 = (const float*)d_in[23];
  const float* ff_w2 = (const float*)d_in[24];
  const float* ff_b2 = (const float*)d_in[25];
  const float* ln2_g = (const float*)d_in[26];
  const float* ln2_b = (const float*)d_in[27];
  const float* skip_w = (const float*)d_in[28];
  const float* skip_b = (const float*)d_in[29];
  const float* proj_out_w = (const float*)d_in[30];
  const float* proj_out_b = (const float*)d_in[31];

  char* p = (char*)d_ws;
  auto alloc = [&](size_t bytes) -> void* {
    void* r = (void*)p;
    p += (bytes + 255) & ~(size_t)255;
    return r;
  };
  // activations / staging
  uint16_t* hin_bf = (uint16_t*)alloc((size_t)cBT * cCin * 2);
  float*    h_f    = (float*)   alloc((size_t)cBT * cD * 4);
  uint16_t* h_bf   = (uint16_t*)alloc((size_t)cBT * cD * 2);
  float*    t32a   = (float*)   alloc((size_t)cBT * cD * 4);
  float*    qf     = (float*)   alloc((size_t)cBT * cD * 4);
  float*    kf     = (float*)   alloc((size_t)cBT * cD * 4);
  float*    vf     = (float*)   alloc((size_t)cBT * cD * 4);
  uint16_t* Qbf    = (uint16_t*)alloc((size_t)cBT * cD * 2);
  uint16_t* Kbf    = (uint16_t*)alloc((size_t)cBT * cD * 2);
  uint16_t* VTbf   = (uint16_t*)alloc((size_t)cBT * cD * 2);
  uint16_t* Obf    = (uint16_t*)alloc((size_t)cBT * cD * 2);
  uint16_t* ffbf   = (uint16_t*)alloc((size_t)cBT * cF * 2);
  uint16_t* catbf  = (uint16_t*)alloc((size_t)cBT * 2 * cD * 2);
  uint16_t* skipbf[4];
  for (int i = 0; i < 4; ++i) skipbf[i] = (uint16_t*)alloc((size_t)cBT * cD * 2);
  float*    tmpo   = (float*)   alloc((size_t)cBT * cNF * 4);
  // bf16 weight pool (converted once per launch; ~28us of HBM traffic vs WMMA-bound math)
  const size_t DD = (size_t)cD * cD;
  uint16_t* w_pin  = (uint16_t*)alloc((size_t)cD * cCin * 2);
  uint16_t* w_q    = (uint16_t*)alloc((size_t)cL * DD * 2);
  uint16_t* w_k    = (uint16_t*)alloc((size_t)cL * DD * 2);
  uint16_t* w_v    = (uint16_t*)alloc((size_t)cL * DD * 2);
  uint16_t* w_o    = (uint16_t*)alloc((size_t)cL * DD * 2);
  uint16_t* w_f1   = (uint16_t*)alloc((size_t)cL * cF * cD * 2);
  uint16_t* w_f2   = (uint16_t*)alloc((size_t)cL * cD * cF * 2);
  uint16_t* w_sk   = (uint16_t*)alloc((size_t)(cL / 2) * cD * 2 * cD * 2);
  uint16_t* w_pout = (uint16_t*)alloc((size_t)cNF * cD * 2);

  auto cvt = [&](const float* s, uint16_t* d, size_t n) {
    k_cvt<<<4096, 256, 0, stream>>>(s, d, (int)n);
  };
  cvt(proj_in_w, w_pin, (size_t)cD * cCin);
  cvt(qw, w_q, (size_t)cL * DD);
  cvt(kw, w_k, (size_t)cL * DD);
  cvt(vw, w_v, (size_t)cL * DD);
  cvt(ow, w_o, (size_t)cL * DD);
  cvt(ff_w1, w_f1, (size_t)cL * cF * cD);
  cvt(ff_w2, w_f2, (size_t)cL * cD * cF);
  cvt(skip_w, w_sk, (size_t)(cL / 2) * cD * 2 * cD);
  cvt(proj_out_w, w_pout, (size_t)cNF * cD);

  auto gemm = [&](const uint16_t* A, const uint16_t* W, const float* bias,
                  float* Cf, uint16_t* Cbf, int M, int N, int Kd, float alpha, int act) {
    dim3 grid(M / 64, (N + 127) / 128);
    k_gemm<<<grid, 256, 0, stream>>>(A, W, bias, Cf, Cbf, M, N, Kd, alpha, act);
  };

  // ---- front end ----
  k_build_hin<<<2048, 256, 0, stream>>>(x, mu, tokens, emb, hin_bf);
  gemm(hin_bf, w_pin, proj_in_b, h_f, nullptr, cBT, cD, cCin, 1.0f, 0);
  k_temb<<<4096, 256, 0, stream>>>(h_f, tvals, ymask);
  for (int dconv = 0; dconv < 2; ++dconv) {
    k_conv<<<cB * cG * 32, 256, 0, stream>>>(
        h_f, posconv_w + (size_t)dconv * cD * 64 * CONV_K, posconv_b + dconv * cD, ymask, t32a);
    k_add<<<4096, 256, 0, stream>>>(h_f, t32a);
  }
  k_ln<<<cBT, 256, 0, stream>>>(h_f, nullptr, ymask, ln0_g, ln0_b, h_f, h_bf, nullptr, 0);

  // ---- transformer layers ----
  for (int i = 0; i < cL; ++i) {
    if (i >= cL / 2) {
      k_concat<<<4096, 256, 0, stream>>>(h_bf, skipbf[cL - 1 - i], catbf);
      gemm(catbf, w_sk + (size_t)(i - cL / 2) * cD * 2 * cD,
           skip_b + (size_t)(i - cL / 2) * cD, h_f, h_bf, cBT, cD, 2 * cD, 1.0f, 0);
    }
    gemm(h_bf, w_q + (size_t)i * DD, qb + (size_t)i * cD, qf, nullptr, cBT, cD, cD, 0.125f, 0);
    gemm(h_bf, w_k + (size_t)i * DD, kb + (size_t)i * cD, kf, nullptr, cBT, cD, cD, 1.0f, 0);
    gemm(h_bf, w_v + (size_t)i * DD, vb + (size_t)i * cD, vf, nullptr, cBT, cD, cD, 1.0f, 0);
    k_pack_qkv<<<4096, 256, 0, stream>>>(qf, kf, vf, Qbf, Kbf, VTbf);
    k_attn<<<cB * cH * (cT / 64), 256, (size_t)64 * SW * 4, stream>>>(Qbf, Kbf, VTbf, ymask, Obf);
    gemm(Obf, w_o + (size_t)i * DD, ob + (size_t)i * cD, t32a, nullptr, cBT, cD, cD, 1.0f, 0);
    k_ln<<<cBT, 256, 0, stream>>>(h_f, t32a, ymask, ln1_g + (size_t)i * cD, ln1_b + (size_t)i * cD,
                                  h_f, h_bf, nullptr, 1 | 2);
    gemm(h_bf, w_f1 + (size_t)i * cF * cD, ff_b1 + (size_t)i * cF, nullptr, ffbf,
         cBT, cF, cD, 1.0f, 1);
    gemm(ffbf, w_f2 + (size_t)i * cD * cF, ff_b2 + (size_t)i * cD, t32a, nullptr,
         cBT, cD, cF, 1.0f, 0);
    k_ln<<<cBT, 256, 0, stream>>>(h_f, t32a, ymask, ln2_g + (size_t)i * cD, ln2_b + (size_t)i * cD,
                                  h_f, h_bf, (i < cL / 2) ? skipbf[i] : nullptr, 1 | 2 | 4);
  }

  // ---- output projection ----
  gemm(h_bf, w_pout, proj_out_b, tmpo, nullptr, cBT, cNF, cD, 1.0f, 0);
  k_outT<<<1280, 256, 0, stream>>>(tmpo, ymask, (float*)d_out);
}
